// STRG_3289944949164
// MI455X (gfx1250) — matrix operand order repaired and checked by reading, not compiled, not run on hardware
//
#include <hip/hip_runtime.h>

#define B_ 64
#define C_ 128
#define T_ 2048
#define F_ 5
#define N_ (C_ * F_)            // 640
#define LDS_STRIDE 40           // halves per tile row (32 data + 8 pad -> 80B, 8B aligned)
#define TILE_H (C_ * LDS_STRIDE)

typedef __attribute__((ext_vector_type(16))) _Float16 v16h;
typedef __attribute__((ext_vector_type(8)))  float    v8f;

union Frag16 { v16h v; uint2 u[4]; };
union Pack4  { _Float16 h[4]; uint2 u; };

// ---------------------------------------------------------------------------
// Kernel 1: zero-fill the A_norm region (off-block entries must be 0; block
// diagonal gets overwritten by the fused kernel afterwards on the same stream).
// ---------------------------------------------------------------------------
__global__ __launch_bounds__(256) void strg_zero(float4* __restrict__ out, int n4) {
    int i = blockIdx.x * 256 + threadIdx.x;
    if (i < n4) out[i] = make_float4(0.f, 0.f, 0.f, 0.f);
}

// ---------------------------------------------------------------------------
// Kernel 2: single-pass fused kernel for one (batch, band) 128x128 block:
//   raw-data GEMM  G = X * X^T  via v_wmma_f32_16x16x32_f16 (hi+lo f16 split),
//   per-row sum/sumsq accumulated during tile loads,
//   epilogue: cov fix-up -> |corr| -> 0.5*spatial + 0.5*func -> D^-1/2 A D^-1/2.
// Ping-pong LDS tiles (one barrier per K-chunk) + register-double-buffered
// global prefetch so loads overlap the WMMA stream.
// ---------------------------------------------------------------------------
__global__ __launch_bounds__(256) void strg_fused(
    const float* __restrict__ p0, const float* __restrict__ p1,
    const float* __restrict__ p2, const float* __restrict__ p3,
    const float* __restrict__ p4, float* __restrict__ out)
{
    __shared__ _Float16 tileHi[2 * TILE_H];
    __shared__ _Float16 tileLo[2 * TILE_H];
    __shared__ float partS[256];
    __shared__ float partS2[256];
    __shared__ float meanS[C_];
    __shared__ float rinvS[C_];
    __shared__ float rowsum[C_];
    __shared__ float dinv[C_];

    const int f = blockIdx.x;       // band
    const int b = blockIdx.y;       // batch
    const float* X = (f == 0) ? p0 : (f == 1) ? p1 : (f == 2) ? p2 : (f == 3) ? p3 : p4;
    X += (size_t)b * C_ * T_;

    const int tid  = threadIdx.x;
    const int wave = tid >> 5;
    const int lane = tid & 31;

    const int r2 = tid >> 1;        // tile row this thread loads (0..127)
    const int h2 = tid & 1;         // which 16-float half of the 32-wide chunk

    const int mrow  = wave * 16 + (lane & 15);  // A fragment row
    const int kb    = (lane >> 4) * 8;          // A fragment K sub-base
    const int lcol  = lane & 15;
    const int kh    = lane >> 4;                // B fragment K half
    const int lhalf = lane >> 4;

    v8f acc[8];
    #pragma unroll
    for (int j = 0; j < 8; ++j)
        #pragma unroll
        for (int e = 0; e < 8; ++e) acc[j][e] = 0.0f;

    float s = 0.f, s2 = 0.f;

    // -------- K loop: raw tiles, stats on the fly, ping-pong LDS, WMMA ------
    const float4* srcBase = (const float4*)(X + (size_t)r2 * T_);
    float4 rg[2][4];
    #pragma unroll
    for (int q = 0; q < 4; ++q) rg[0][q] = srcBase[h2 * 4 + q];

    for (int kc = 0; kc < T_; kc += 32) {
        const int pb = (kc >> 5) & 1;

        // prefetch next chunk into the other register set
        if (kc + 32 < T_) {
            #pragma unroll
            for (int q = 0; q < 4; ++q)
                rg[pb ^ 1][q] = srcBase[((kc + 32) >> 2) + h2 * 4 + q];
        }

        _Float16* tH = tileHi + pb * TILE_H;
        _Float16* tL = tileLo + pb * TILE_H;

        // consume current chunk: stats + f16 hi/lo split into LDS
        #pragma unroll
        for (int q = 0; q < 4; ++q) {
            float4 v = rg[pb][q];
            s  += v.x + v.y + v.z + v.w;
            s2 += v.x * v.x + v.y * v.y + v.z * v.z + v.w * v.w;
            Pack4 hi, lo;
            hi.h[0] = (_Float16)v.x; lo.h[0] = (_Float16)(v.x - (float)hi.h[0]);
            hi.h[1] = (_Float16)v.y; lo.h[1] = (_Float16)(v.y - (float)hi.h[1]);
            hi.h[2] = (_Float16)v.z; lo.h[2] = (_Float16)(v.z - (float)hi.h[2]);
            hi.h[3] = (_Float16)v.w; lo.h[3] = (_Float16)(v.w - (float)hi.h[3]);
            const int off = r2 * LDS_STRIDE + h2 * 16 + q * 4;
            *(uint2*)&tH[off] = hi.u;
            *(uint2*)&tL[off] = lo.u;
        }
        __syncthreads();   // includes s_wait_dscnt 0: prior reads+stores done

        // A fragments (16x32): lane (m + 16*g) holds row m, K = g*8 + {0..7,16..23}
        Frag16 aH, aL;
        {
            const int base = mrow * LDS_STRIDE;
            aH.u[0] = *(uint2*)&tH[base + kb];
            aH.u[1] = *(uint2*)&tH[base + kb + 4];
            aH.u[2] = *(uint2*)&tH[base + 16 + kb];
            aH.u[3] = *(uint2*)&tH[base + 16 + kb + 4];
            aL.u[0] = *(uint2*)&tL[base + kb];
            aL.u[1] = *(uint2*)&tL[base + kb + 4];
            aL.u[2] = *(uint2*)&tL[base + 16 + kb];
            aL.u[3] = *(uint2*)&tL[base + 16 + kb + 4];
        }

        #pragma unroll
        for (int j = 0; j < 8; ++j) {
            // B fragment (32x16): lane (n + 16*kh) holds col n, K = kh*8 + {0..7,16..23}
            const int base = (j * 16 + lcol) * LDS_STRIDE + kh * 8;
            Frag16 bH, bL;
            bH.u[0] = *(uint2*)&tH[base];
            bH.u[1] = *(uint2*)&tH[base + 4];
            bH.u[2] = *(uint2*)&tH[base + 16];
            bH.u[3] = *(uint2*)&tH[base + 20];
            bL.u[0] = *(uint2*)&tL[base];
            bL.u[1] = *(uint2*)&tL[base + 4];
            bL.u[2] = *(uint2*)&tL[base + 16];
            bL.u[3] = *(uint2*)&tL[base + 20];

            acc[j] = __builtin_amdgcn_wmma_f32_16x16x32_f16(
                false, aH.v, false, bH.v, (short)0, acc[j], false, false);
            acc[j] = __builtin_amdgcn_wmma_f32_16x16x32_f16(
                false, aH.v, false, bL.v, (short)0, acc[j], false, false);
            acc[j] = __builtin_amdgcn_wmma_f32_16x16x32_f16(
                false, aL.v, false, bH.v, (short)0, acc[j], false, false);
        }
        // no trailing barrier: next iteration stores the other buffer; the
        // next __syncthreads (with its dscnt wait) protects re-use.
    }

    // -------- per-row statistics reduction ----------------------------------
    partS[tid]  = s;
    partS2[tid] = s2;
    __syncthreads();
    if (tid < C_) {
        float ts  = partS[2 * tid] + partS[2 * tid + 1];
        float ts2 = partS2[2 * tid] + partS2[2 * tid + 1];
        float mean = ts * (1.0f / T_);
        float var  = fmaxf((ts2 - ts * mean) * (1.0f / (T_ - 1)), 0.0f);
        float rinv = 1.0f / (sqrtf(var) + 1e-8f);
        meanS[tid] = mean;
        rinvS[tid] = rinv;
        // node_features[b, f*C + tid] = var
        out[(size_t)N_ * N_ * B_ + (size_t)b * N_ + f * C_ + tid] = var;
        // bandpowers[b, tid, f] = var
        out[(size_t)N_ * N_ * B_ + (size_t)B_ * N_ + (size_t)b * C_ * F_ + tid * F_ + f] = var;
    }
    __syncthreads();

    // -------- epilogue: cov fix-up, blend, degree normalization -------------
    // C/D layout: lane l, VGPR p -> row = p + 8*(l>=16), col = l&15 (per tile j)
    float mu_r[8], ri_r[8];
    #pragma unroll
    for (int p = 0; p < 8; ++p) {
        const int r = wave * 16 + lhalf * 8 + p;
        mu_r[p] = meanS[r];
        ri_r[p] = rinvS[r];
    }

    float rs[8];
    #pragma unroll
    for (int p = 0; p < 8; ++p) rs[p] = 0.f;

    #pragma unroll
    for (int j = 0; j < 8; ++j) {
        const int c = j * 16 + lcol;
        const float mu_c = meanS[c];
        const float ri_c = rinvS[c];
        #pragma unroll
        for (int p = 0; p < 8; ++p) {
            const int r = wave * 16 + lhalf * 8 + p;
            const float cov = acc[j][p] - (float)T_ * mu_r[p] * mu_c;
            const float fc  = (r == c) ? 1.0f
                              : fabsf(cov) * ri_r[p] * ri_c * (1.0f / (T_ - 1));
            int d = r - c; if (d < 0) d = -d;
            const float sp = (d >= 1 && d <= 3) ? 1.0f : 0.0f;
            const float val = 0.5f * sp + 0.5f * fc;
            acc[j][p] = val;
            float t = val;                 // reduce across 16 columns of this half
            t += __shfl_xor(t, 1, 32);
            t += __shfl_xor(t, 2, 32);
            t += __shfl_xor(t, 4, 32);
            t += __shfl_xor(t, 8, 32);
            rs[p] += t;
        }
    }
    if (lcol == 0) {
        #pragma unroll
        for (int p = 0; p < 8; ++p)
            rowsum[wave * 16 + lhalf * 8 + p] = rs[p];
    }
    __syncthreads();
    if (tid < C_) dinv[tid] = rsqrtf(rowsum[tid] + 1e-8f);
    __syncthreads();

    float* outA = out + (size_t)b * N_ * N_;
    #pragma unroll
    for (int j = 0; j < 8; ++j) {
        const int c = j * 16 + lcol;
        const float dc = dinv[c];
        #pragma unroll
        for (int p = 0; p < 8; ++p) {
            const int r = wave * 16 + lhalf * 8 + p;
            outA[(size_t)(f * C_ + r) * N_ + f * C_ + c] = acc[j][p] * dinv[r] * dc;
        }
    }
}

// ---------------------------------------------------------------------------
extern "C" void kernel_launch(void* const* d_in, const int* in_sizes, int n_in,
                              void* d_out, int out_size, void* d_ws, size_t ws_size,
                              hipStream_t stream) {
    (void)in_sizes; (void)n_in; (void)out_size; (void)d_ws; (void)ws_size;
    const float* p0 = (const float*)d_in[0];   // delta
    const float* p1 = (const float*)d_in[1];   // theta
    const float* p2 = (const float*)d_in[2];   // alpha
    const float* p3 = (const float*)d_in[3];   // beta
    const float* p4 = (const float*)d_in[4];   // gamma
    float* out = (float*)d_out;

    const int n4 = (N_ * N_ * B_) / 4;         // A_norm region as float4
    strg_zero<<<dim3((n4 + 255) / 256), dim3(256), 0, stream>>>((float4*)out, n4);
    strg_fused<<<dim3(F_, B_), dim3(256), 0, stream>>>(p0, p1, p2, p3, p4, out);
}
